// PAN_86586540687832
// MI455X (gfx1250) — matrix operand order repaired and checked by reading, not compiled, not run on hardware
//
#include <hip/hip_runtime.h>
#include <hip/hip_bf16.h>

typedef __attribute__((ext_vector_type(16))) __bf16 v16bf;
typedef __attribute__((ext_vector_type(8)))  __bf16 v8bf;
typedef __attribute__((ext_vector_type(8)))  float  v8f;
typedef __attribute__((ext_vector_type(4)))  int    v4i;

#define N_NODES 8192
#define DF      128
#define S_SUP   2
#define L_LAY   2
#define E_EDGE  (1 << 20)
#define NOUT    8
#define BN_EPS  1e-5f

// GEMM tiling: 256 threads = 8 waves. Block tile = 32 (M) x 128 (N).
// K-stage = 64 (two 16x16x32 WMMA k-chunks), double-buffered LDS.
#define TM   32
#define TK   64
#define ATR  72   // padded LDS row length in halfs (144B = 9x16B, keeps b128 alignment)

#if __has_builtin(__builtin_amdgcn_global_load_async_to_lds_b128) && \
    __has_builtin(__builtin_amdgcn_s_wait_asynccnt)
#define USE_ASYNC 1
#else
#define USE_ASYNC 0
#endif

// global (addrspace 1) / LDS (addrspace 3) typed vector pointers for the
// async-to-LDS builtin: signature wants int4* in the matching address spaces.
#define GASV(p) ((__attribute__((address_space(1))) v4i*)(p))
#define LASV(p) ((__attribute__((address_space(3))) v4i*)(p))

__device__ __forceinline__ v16bf cat16(v8bf a, v8bf b) {
  return __builtin_shufflevector(a, b, 0,1,2,3,4,5,6,7,8,9,10,11,12,13,14,15);
}

// C = A(fp32) @ B(split-bf16, stored N-major: B[n][k]) with fp32 accum via
// 3x v_wmma_f32_16x16x32_bf16 (hi*hi + hi*lo + lo*hi)  -> ~fp32 accuracy.
// EPI==0: write C as split-bf16 *transposed* (out[n][m]) -> feeds next GEMM's B.
// EPI==1: ReLU, store fp32 h, accumulate per-column sum/sumsq for BatchNorm.
template<int EPI>
__global__ __launch_bounds__(256)
void gcn_gemm(const float* __restrict__ Abase, long long lda,
              const __bf16* __restrict__ BhBase, const __bf16* __restrict__ BlBase,
              int Ktot, int nseg, long long segA, long long segB,
              long long yB, long long yO,
              float* __restrict__ hOut,
              __bf16* __restrict__ xwTh, __bf16* __restrict__ xwTl,
              float* __restrict__ colsum, float* __restrict__ colsq)
{
  __shared__ __bf16 AhS[2][TM][ATR], AlS[2][TM][ATR];
  __shared__ __bf16 BhS[2][DF][ATR], BlS[2][DF][ATR];
  __shared__ float lsum[DF], lsq[DF];

  const int tid  = threadIdx.x;
  const int wave = tid >> 5, lane = tid & 31;
  const int m0   = blockIdx.x * TM;

  const __bf16* Bh0 = BhBase + (long long)blockIdx.y * yB;
  const __bf16* Bl0 = BlBase + (long long)blockIdx.y * yB;

  if (EPI == 1 && tid < DF) { lsum[tid] = 0.f; lsq[tid] = 0.f; }

  // wave -> (16 rows, 32 cols): 2 n-tiles per wave
  const int mrow = (wave & 1) * 16;
  const int nc0  = (wave >> 1) * 32;

  v8f acc[2];
  #pragma unroll
  for (int t = 0; t < 2; ++t)
    #pragma unroll
    for (int i = 0; i < 8; ++i) acc[t][i] = 0.f;

  const int sps = Ktot / TK;       // stages per segment
  const int nst = nseg * sps;      // total K stages

  // Stage producer: A tile 32x64 fp32 -> split bf16 hi/lo via VALU + ds_store;
  // B tiles 128x64 halfs x2 parts via async DMA straight into LDS (ASYNCcnt).
  auto produce = [&](int buf, int seg, int k0) {
    const float*  A  = Abase + (long long)seg * segA;
    const __bf16* BH = Bh0 + (long long)seg * segB;
    const __bf16* BL = Bl0 + (long long)seg * segB;
    {
      const int m = tid >> 3, kq = (tid & 7) * 8;
      const float4* ap = (const float4*)(A + (long long)(m0 + m) * lda + k0 + kq);
      const float4 u = ap[0], v = ap[1];
      const float f[8] = {u.x, u.y, u.z, u.w, v.x, v.y, v.z, v.w};
      v8bf hv, lv;
      #pragma unroll
      for (int i = 0; i < 8; ++i) {
        hv[i] = (__bf16)f[i];
        lv[i] = (__bf16)(f[i] - (float)hv[i]);
      }
      *(v8bf*)&AhS[buf][m][kq] = hv;
      *(v8bf*)&AlS[buf][m][kq] = lv;
    }
    {
      const int d = tid >> 1, kb = (tid & 1) * 32;
      const __bf16* gh = BH + (long long)d * Ktot + k0 + kb;
      const __bf16* gl = BL + (long long)d * Ktot + k0 + kb;
      __bf16* lh = &BhS[buf][d][kb];
      __bf16* ll = &BlS[buf][d][kb];
#if USE_ASYNC
      #pragma unroll
      for (int c = 0; c < 4; ++c) {
        __builtin_amdgcn_global_load_async_to_lds_b128(GASV(gh + c * 8), LASV(lh + c * 8), 0, 0);
        __builtin_amdgcn_global_load_async_to_lds_b128(GASV(gl + c * 8), LASV(ll + c * 8), 0, 0);
      }
#else
      #pragma unroll
      for (int c = 0; c < 4; ++c) {
        *(uint4*)(lh + c * 8) = *(const uint4*)(gh + c * 8);
        *(uint4*)(ll + c * 8) = *(const uint4*)(gl + c * 8);
      }
#endif
    }
  };

  int pseg = 0, pk0 = 0;
  produce(0, pseg, pk0);
  pk0 += TK; if (pk0 == Ktot) { pk0 = 0; ++pseg; }

  for (int i = 0; i < nst; ++i) {
    const int p = i & 1;
#if USE_ASYNC
    __builtin_amdgcn_s_wait_asynccnt(0);   // our DMA into buf p has landed in LDS
#endif
    __syncthreads();                        // buf p complete; buf 1-p readers done
    if (i + 1 < nst) {                      // start DMA for next stage into buf 1-p
      produce(1 - p, pseg, pk0);
      pk0 += TK; if (pk0 == Ktot) { pk0 = 0; ++pseg; }
    }
    // ---- compute stage i from buf p: 2 k-chunks x 2 n-tiles x 3 WMMA ----
    #pragma unroll
    for (int kc = 0; kc < 2; ++kc) {
      const int ks  = kc * 32;
      const int am  = mrow + (lane & 15);
      const int aks = ks + (lane >> 4) * 8;      // ISA 16-bit A 16x32 layout
      v16bf ahi = cat16(*(const v8bf*)&AhS[p][am][aks],
                        *(const v8bf*)&AhS[p][am][aks + 16]);
      v16bf alo = cat16(*(const v8bf*)&AlS[p][am][aks],
                        *(const v8bf*)&AlS[p][am][aks + 16]);
      #pragma unroll
      for (int nt = 0; nt < 2; ++nt) {
        const int bd  = nc0 + nt * 16 + (lane & 15);
        const int bks = ks + (lane >> 4) * 8;    // B: same k layout as A (lanes0-15 K0..7,16..23; lanes16-31 K8..15,24..31)
        v16bf bhi = cat16(*(const v8bf*)&BhS[p][bd][bks],
                          *(const v8bf*)&BhS[p][bd][bks + 16]);
        v16bf blo = cat16(*(const v8bf*)&BlS[p][bd][bks],
                          *(const v8bf*)&BlS[p][bd][bks + 16]);
        acc[nt] = __builtin_amdgcn_wmma_f32_16x16x32_bf16(false, ahi, false, bhi,
                                                          (short)0, acc[nt], false, false);
        acc[nt] = __builtin_amdgcn_wmma_f32_16x16x32_bf16(false, ahi, false, blo,
                                                          (short)0, acc[nt], false, false);
        acc[nt] = __builtin_amdgcn_wmma_f32_16x16x32_bf16(false, alo, false, bhi,
                                                          (short)0, acc[nt], false, false);
      }
    }
  }

  // ---- epilogue ----
  if (EPI == 0) {
    #pragma unroll
    for (int nt = 0; nt < 2; ++nt) {
      const int e = nc0 + nt * 16 + (lane & 15);
      const long long mb = m0 + mrow + (lane >> 4) * 8;
      v8bf hv, lv;
      #pragma unroll
      for (int r = 0; r < 8; ++r) {
        const float f = acc[nt][r];
        hv[r] = (__bf16)f;
        lv[r] = (__bf16)(f - (float)hv[r]);
      }
      *(v8bf*)(xwTh + (long long)blockIdx.y * yO + (long long)e * N_NODES + mb) = hv;
      *(v8bf*)(xwTl + (long long)blockIdx.y * yO + (long long)e * N_NODES + mb) = lv;
    }
  } else {
    #pragma unroll
    for (int nt = 0; nt < 2; ++nt) {
      const int n  = nc0 + nt * 16 + (lane & 15);
      const int mb = m0 + mrow + (lane >> 4) * 8;
      float s1 = 0.f, s2 = 0.f;
      #pragma unroll
      for (int r = 0; r < 8; ++r) {
        float f = acc[nt][r];
        f = f > 0.f ? f : 0.f;
        hOut[(long long)(mb + r) * DF + n] = f;
        s1 += f; s2 += f * f;
      }
      atomicAdd(&lsum[n], s1);
      atomicAdd(&lsq[n],  s2);
    }
    __syncthreads();
    if (tid < DF) {
      atomicAdd(&colsum[tid], lsum[tid]);
      atomicAdd(&colsq[tid],  lsq[tid]);
    }
  }
}

// Split W[l,s] (fp32, [d_in][d_out]) into bf16 hi/lo, transposed to [d_out][d_in].
__global__ void split_w(const float* __restrict__ W,
                        __bf16* __restrict__ WTh, __bf16* __restrict__ WTl) {
  const int idx = blockIdx.x * 256 + threadIdx.x;
  if (idx >= L_LAY * S_SUP * DF * DF) return;
  const int e  = idx & 127;
  const int d  = (idx >> 7) & 127;
  const int ls = idx >> 14;
  const float f = W[idx];
  const __bf16 h = (__bf16)f;
  const long long o = ((long long)ls * DF + e) * DF + d;
  WTh[o] = h;
  WTl[o] = (__bf16)(f - (float)h);
}

__global__ void zero_stats(float* colsum, float* colsq) {
  const int t = threadIdx.x;
  if (t < DF) { colsum[t] = 0.f; colsq[t] = 0.f; }
}

__global__ void bn_finalize(const float* __restrict__ colsum, const float* __restrict__ colsq,
                            const float* __restrict__ gamma, const float* __restrict__ beta,
                            float* __restrict__ scale, float* __restrict__ shift) {
  const int d = threadIdx.x;
  if (d >= DF) return;
  const float mu  = colsum[d] * (1.f / (float)N_NODES);
  const float var = colsq[d] * (1.f / (float)N_NODES) - mu * mu;
  const float sc  = gamma[d] * rsqrtf(var + BN_EPS);
  scale[d] = sc;
  shift[d] = beta[d] - mu * sc;
}

__global__ void bn_apply(const float* __restrict__ h, const float* __restrict__ scale,
                         const float* __restrict__ shift, float* __restrict__ xout) {
  const int idx = blockIdx.x * 256 + threadIdx.x;
  if (idx >= N_NODES * DF) return;
  const int d = idx & (DF - 1);
  xout[idx] = scale[d] * h[idx] + shift[d];
}

// Edge decoder: jfeat = |x[r]-x[c]|; sigmoid(fc1)+clip, softmax(fc2), weighted sum.
// x (4MB) is L2-resident; weights go through wave-uniform scalar loads.
__global__ __launch_bounds__(256)
void decoder(const float* __restrict__ x,
             const int* __restrict__ rows, const int* __restrict__ cols,
             const float* __restrict__ fc1w, const float* __restrict__ fc1b,
             const float* __restrict__ fc2w, const float* __restrict__ fc2b,
             float* __restrict__ out, float* __restrict__ attrs, float* __restrict__ imps)
{
  const int e = blockIdx.x * 256 + threadIdx.x;
  if (e >= E_EDGE) return;
  const float4* pr = (const float4*)(x + (long long)rows[e] * DF);
  const float4* pc = (const float4*)(x + (long long)cols[e] * DF);
  const float4* w1 = (const float4*)fc1w;   // uniform addresses -> s_load
  const float4* w2 = (const float4*)fc2w;

  float a1[NOUT], a2[NOUT];
  #pragma unroll
  for (int o = 0; o < NOUT; ++o) { a1[o] = 0.f; a2[o] = 0.f; }

  for (int k = 0; k < DF / 4; ++k) {
    const float4 u = pr[k], v = pc[k];
    const float jx = fabsf(u.x - v.x), jy = fabsf(u.y - v.y);
    const float jz = fabsf(u.z - v.z), jw = fabsf(u.w - v.w);
    #pragma unroll
    for (int o = 0; o < NOUT; ++o) {
      const float4 q1 = w1[o * (DF / 4) + k];
      a1[o] += jx * q1.x + jy * q1.y + jz * q1.z + jw * q1.w;
      const float4 q2 = w2[o * (DF / 4) + k];
      a2[o] += jx * q2.x + jy * q2.y + jz * q2.z + jw * q2.w;
    }
  }

  float at[NOUT], mx = -3.4e38f;
  #pragma unroll
  for (int o = 0; o < NOUT; ++o) {
    const float z = a1[o] + fc1b[o];
    const float sg = 1.f / (1.f + __expf(-z));
    at[o] = fminf(fmaxf(sg, 0.f), 1.f);
    a2[o] += fc2b[o];
    mx = fmaxf(mx, a2[o]);
  }
  float ex[NOUT], den = 0.f;
  #pragma unroll
  for (int o = 0; o < NOUT; ++o) { ex[o] = __expf(a2[o] - mx); den += ex[o]; }
  const float inv = 1.f / den;
  float accv = 0.f;
  float im[NOUT];
  #pragma unroll
  for (int o = 0; o < NOUT; ++o) { im[o] = ex[o] * inv; accv += at[o] * im[o]; }

  out[e] = fminf(fmaxf(accv, 0.f), 1.f);
  float4* ao = (float4*)(attrs + (long long)e * NOUT);
  ao[0] = make_float4(at[0], at[1], at[2], at[3]);
  ao[1] = make_float4(at[4], at[5], at[6], at[7]);
  float4* io = (float4*)(imps + (long long)e * NOUT);
  io[0] = make_float4(im[0], im[1], im[2], im[3]);
  io[1] = make_float4(im[4], im[5], im[6], im[7]);
}

extern "C" void kernel_launch(void* const* d_in, const int* in_sizes, int n_in,
                              void* d_out, int out_size, void* d_ws, size_t ws_size,
                              hipStream_t stream) {
  const float* x0    = (const float*)d_in[0];
  const float* sup   = (const float*)d_in[1];
  const float* W     = (const float*)d_in[2];
  const float* gamma = (const float*)d_in[3];
  const float* beta  = (const float*)d_in[4];
  const float* fc1w  = (const float*)d_in[5];
  const float* fc1b  = (const float*)d_in[6];
  const float* fc2w  = (const float*)d_in[7];
  const float* fc2b  = (const float*)d_in[8];
  const int*   rows  = (const int*)d_in[9];
  const int*   cols  = (const int*)d_in[10];

  char* ws = (char*)d_ws;
  __bf16* xwTh  = (__bf16*)(ws);                                 // 4 MB  [S][DF][N]
  __bf16* xwTl  = (__bf16*)(ws + (4ll << 20));                   // 4 MB
  float*  hbuf  = (float*)(ws + (8ll << 20));                    // 4 MB  [N][DF]
  float*  xcur  = (float*)(ws + (12ll << 20));                   // 4 MB  [N][DF]
  __bf16* WTh   = (__bf16*)(ws + (16ll << 20));                  // 128 KB
  __bf16* WTl   = (__bf16*)(ws + (16ll << 20) + (128 << 10));    // 128 KB
  float*  colsum= (float*)(ws + (16ll << 20) + (256 << 10));
  float*  colsq = colsum + 128;
  float*  scale = colsum + 256;
  float*  shift = colsum + 384;

  split_w<<<(L_LAY * S_SUP * DF * DF + 255) / 256, 256, 0, stream>>>(W, WTh, WTl);

  const int MT = N_NODES / TM;  // 256 M-tiles
  for (int l = 0; l < L_LAY; ++l) {
    const float* xin = (l == 0) ? x0 : xcur;
    zero_stats<<<1, 256, 0, stream>>>(colsum, colsq);
    // GEMM1: xwT[s] = split_bf16((xin @ W[l,s])^T), grid.y = s
    gcn_gemm<0><<<dim3(MT, S_SUP), 256, 0, stream>>>(
        xin, (long long)DF,
        WTh + (long long)l * S_SUP * DF * DF, WTl + (long long)l * S_SUP * DF * DF,
        DF, 1, 0ll, 0ll,
        /*yB*/(long long)DF * DF, /*yO*/(long long)DF * N_NODES,
        nullptr, xwTh, xwTl, nullptr, nullptr);
    // GEMM2: h = relu(sum_s supports[s] @ xw[s]) + BN stats
    gcn_gemm<1><<<dim3(MT, 1), 256, 0, stream>>>(
        sup, (long long)N_NODES, xwTh, xwTl,
        N_NODES, S_SUP,
        (long long)N_NODES * N_NODES, (long long)DF * N_NODES,
        0ll, 0ll,
        hbuf, nullptr, nullptr, colsum, colsq);
    bn_finalize<<<1, 128, 0, stream>>>(colsum, colsq, gamma + l * DF, beta + l * DF,
                                       scale, shift);
    bn_apply<<<(N_NODES * DF + 255) / 256, 256, 0, stream>>>(hbuf, scale, shift, xcur);
  }

  float* outp  = (float*)d_out;
  float* attrs = outp + E_EDGE;
  float* impsp = attrs + (long long)E_EDGE * NOUT;
  decoder<<<(E_EDGE + 255) / 256, 256, 0, stream>>>(xcur, rows, cols,
                                                    fc1w, fc1b, fc2w, fc2b,
                                                    outp, attrs, impsp);
}
